// TransformerPointNetSetAbstractionMsg_17540646436963
// MI455X (gfx1250) — matrix operand order repaired and checked by reading, not compiled, not run on hardware
//
#include <hip/hip_runtime.h>
#include <hip/hip_bf16.h>

// ---------------------------------------------------------------------------
// TransformerPointNetSetAbstractionMsg for MI455X (gfx1250, wave32, WMMA).
//
// Input leaf order assumption (jax tree flatten, dict keys sorted inside
// params; top-level dict in insertion order):
//   0: xyz (16,4096,3)  1: features (16,4096,6)
//   per scale s, base = 2 + 30*s:
//     +0..2  convB[0..2]
//     +3..5  convW[0..2]          (shape (co,ci))
//     +6..17 layer0 {W1,W2,Wk,Wo,Wq,Wv,b1,b2,ln1_b,ln1_g,ln2_b,ln2_g}
//     +18..29 layer1 (same order)
// Output: new_xyz (16,1024,3) fp32 then features (16,1024,768) fp32, flat.
//
// Weights are N-major ([N][K] f16): each lane's B fragment is two 8-element K
// runs, 16 apart (2 x b128). GEMM tiles stream through LDS with double-buffered
// async global->LDS copies (ASYNCcnt pipeline).
// ---------------------------------------------------------------------------

typedef __attribute__((ext_vector_type(16))) _Float16 v16h;
typedef __attribute__((ext_vector_type(8)))  float    v8f;

#define BB 16
#define NN 4096
#define SS 1024
#define DMODEL 128

struct alignas(16) H8 { _Float16 h[8]; };

__device__ inline v8f vzero8() {
  v8f z;
#pragma unroll
  for (int i = 0; i < 8; ++i) z[i] = 0.f;
  return z;
}

__device__ inline v8f wmma_f16(v16h a, v16h b, v8f c) {
  return __builtin_amdgcn_wmma_f32_16x16x32_f16(false, a, false, b, (short)0, c,
                                                false, false);
}

__device__ inline void ld8(v16h& r, int base, const _Float16* p) {
  H8 x = *(const H8*)p;
#pragma unroll
  for (int e = 0; e < 8; ++e) r[base + e] = x.h[e];
}

// A fragment (16x32 f16), row-major activations, leading dim ld.
__device__ inline v16h frag_A(const _Float16* s, int ld, int row0, int k0, int lane) {
  int r  = row0 + (lane & 15);
  int kb = k0 + ((lane >> 4) << 3);
  const _Float16* p = s + (size_t)r * ld + kb;
  v16h a;
  ld8(a, 0, p);
  ld8(a, 8, p + 16);
  return a;
}

// B fragment (32x16 f16) from N-major [N][K] buffer, leading dim ldk.
__device__ inline v16h frag_Bn(const _Float16* s, int ldk, int k0, int n0, int lane) {
  int n  = n0 + (lane & 15);
  int kb = k0 + ((lane >> 4) << 3);
  const _Float16* p = s + (size_t)n * ldk + kb;
  v16h b;
  ld8(b, 0, p);
  ld8(b, 8, p + 16);
  return b;
}

// Issue one async 16-byte global->LDS copy (GVS form, tracked by ASYNCcnt).
__device__ inline void async_b128(const void* lds, unsigned goff,
                                  unsigned long long gbase) {
  unsigned l = (unsigned)(size_t)lds;
  asm volatile("global_load_async_to_lds_b128 %0, %1, %2"
               :: "v"(l), "v"(goff), "s"(gbase) : "memory");
}

// ---------------------------------------------------------------------------
// Farthest point sampling: one block (8 waves) per batch; shfl reductions.
// ---------------------------------------------------------------------------
__global__ void fps_kernel(const float* __restrict__ xyz, float* __restrict__ newXyz) {
  __shared__ float px[NN], py[NN], pz[NN];
  __shared__ float wv[8];
  __shared__ int   wi[8];
  const int b = blockIdx.x, t = threadIdx.x;
  const int wave = t >> 5, lane = t & 31;
  for (int p = t; p < NN; p += 256) {
    const float* q = xyz + (size_t)(b * NN + p) * 3;
    px[p] = q[0]; py[p] = q[1]; pz[p] = q[2];
  }
  float dist[16];
#pragma unroll
  for (int j = 0; j < 16; ++j) dist[j] = 1e10f;
  __syncthreads();
  int cur = 0;
  if (t == 0) {
    float* o = newXyz + (size_t)(b * SS) * 3;
    o[0] = px[0]; o[1] = py[0]; o[2] = pz[0];
  }
  for (int it = 1; it < SS; ++it) {
    const float cx = px[cur], cy = py[cur], cz = pz[cur];
    float bv = -1.f; int bi = 0;
#pragma unroll
    for (int j = 0; j < 16; ++j) {
      int p = j * 256 + t;
      float dx = px[p] - cx, dy = py[p] - cy, dz = pz[p] - cz;
      float d = dx * dx + dy * dy + dz * dz;
      float nd = fminf(dist[j], d);
      dist[j] = nd;
      if (nd > bv) { bv = nd; bi = p; }
    }
    // wave32 argmax reduction
#pragma unroll
    for (int m = 16; m; m >>= 1) {
      float ov = __shfl_xor(bv, m, 32);
      int   oi = __shfl_xor(bi, m, 32);
      if (ov > bv || (ov == bv && oi < bi)) { bv = ov; bi = oi; }
    }
    if (lane == 0) { wv[wave] = bv; wi[wave] = bi; }
    __syncthreads();
    float cv = wv[0]; int ci = wi[0];
#pragma unroll
    for (int k = 1; k < 8; ++k) {
      float v = wv[k]; int ii = wi[k];
      if (v > cv || (v == cv && ii < ci)) { cv = v; ci = ii; }
    }
    cur = ci;
    __syncthreads();
    if (t == 0) {
      float* o = newXyz + (size_t)(b * SS + it) * 3;
      o[0] = px[cur]; o[1] = py[cur]; o[2] = pz[cur];
    }
  }
}

// ---------------------------------------------------------------------------
// Ball query: thread per (b,s); first-K ascending indices within radius.
// ---------------------------------------------------------------------------
__global__ void ball_kernel(const float* __restrict__ xyz,
                            const float* __restrict__ newXyz,
                            int* __restrict__ ball, float r2, int Kn) {
  int q = blockIdx.x * blockDim.x + threadIdx.x;
  if (q >= BB * SS) return;
  int b = q >> 10;
  const float qx = newXyz[q * 3 + 0], qy = newXyz[q * 3 + 1], qz = newXyz[q * 3 + 2];
  const float* base = xyz + (size_t)b * NN * 3;
  int* out = ball + (size_t)q * Kn;
  int cnt = 0;
  for (int n = 0; n < NN && cnt < Kn; ++n) {
    float dx = base[n * 3 + 0] - qx;
    float dy = base[n * 3 + 1] - qy;
    float dz = base[n * 3 + 2] - qz;
    if (dx * dx + dy * dy + dz * dz <= r2) out[cnt++] = n;
  }
  int f = (cnt > 0) ? out[0] : 0;
  for (; cnt < Kn; ++cnt) out[cnt] = f;
}

// ---------------------------------------------------------------------------
// Weight conversion. Conv weights (co,ci) -> f16 [co][cipad] (N-major, padded).
// Transformer weights (K,N) -> f16 [N][K] (transpose-convert).
// ---------------------------------------------------------------------------
__global__ void conv_w_kernel(const float* __restrict__ src, _Float16* __restrict__ dst,
                              int ci, int cipad, int co) {
  int i = blockIdx.x * blockDim.x + threadIdx.x;
  if (i >= cipad * co) return;
  int n = i / cipad, k = i % cipad;
  float v = (k < ci) ? src[(size_t)n * ci + k] : 0.f;
  dst[i] = (_Float16)v;
}

__global__ void cvt_t_kernel(const float* __restrict__ src, _Float16* __restrict__ dst,
                             int K, int N) {
  int i = blockIdx.x * blockDim.x + threadIdx.x;
  if (i >= K * N) return;
  int n = i / K, k = i % K;
  dst[i] = (_Float16)src[(size_t)k * N + n];
}

// V columns of the packed QKV (B,S,384) -> vT[(b*4+h)*32+d][S].
__global__ void vtrans_kernel(const _Float16* __restrict__ qkv, _Float16* __restrict__ vT) {
  int i = blockIdx.x * blockDim.x + threadIdx.x;
  if (i >= BB * SS * 128) return;
  int c = i & 127, q = i >> 7;
  int b = q >> 10, j = q & 1023;
  int h = c >> 5, d = c & 31;
  vT[((size_t)((b * 4 + h) * 32 + d)) * SS + j] = qkv[(size_t)q * 384 + 256 + c];
}

// ---------------------------------------------------------------------------
// Fused grouping + 3-layer MLP + K-maxpool (templated, unrolled WMMA).
// block = 128 threads (4 waves); 64 rows live in LDS across all layers.
// ---------------------------------------------------------------------------
template <int CP, int CO>
__device__ inline void mlp_layer(const _Float16* __restrict__ W,
                                 const float* __restrict__ bias,
                                 const _Float16* cur, _Float16* nxt,
                                 _Float16* Wb, int t, int wave, int lane) {
  const unsigned long long wb64 = (unsigned long long)(size_t)W;
#pragma unroll
  for (int i = t * 8; i < CP * CO; i += 128 * 8)
    async_b128(Wb + i, (unsigned)(i * sizeof(_Float16)), wb64);
  asm volatile("s_wait_asynccnt 0x0" ::: "memory");
  __syncthreads();
#pragma unroll
  for (int nt = 0; nt < CO / 16; ++nt) {
    v8f acc = vzero8();
#pragma unroll
    for (int ks = 0; ks < CP; ks += 32) {
      v16h a = frag_A(cur, 128, wave * 16, ks, lane);
      v16h b = frag_Bn(Wb, CP, ks, nt * 16, lane);
      acc = wmma_f16(a, b, acc);
    }
    const int col = nt * 16 + (lane & 15);
    const float bv = bias[col];
    const int rbase = wave * 16 + ((lane >> 4) << 3);
#pragma unroll
    for (int r8 = 0; r8 < 8; ++r8) {
      float v = acc[r8] + bv;
      v = v > 0.f ? v : 0.f;
      nxt[(rbase + r8) * 128 + col] = (_Float16)v;
    }
  }
  __syncthreads();
}

template <int KN, int SPB, int CP0, int CO0, int CP1, int CO1, int CP2, int CO2,
          int WOFF1, int WOFF2>
__global__ void conv_mlp_kernel(const float* __restrict__ xyz,
                                const float* __restrict__ feat,
                                const float* __restrict__ newXyz,
                                const int* __restrict__ ball,
                                const _Float16* __restrict__ Wall,
                                const float* __restrict__ bias0,
                                const float* __restrict__ bias1,
                                const float* __restrict__ bias2,
                                float* __restrict__ outFeat,
                                float* __restrict__ xcur,
                                _Float16* __restrict__ xf16, int c0) {
  __shared__ __align__(16) _Float16 bufA[64 * 128];
  __shared__ __align__(16) _Float16 bufB[64 * 128];
  __shared__ __align__(16) _Float16 Wb[128 * 128];
  const int t = threadIdx.x, wave = t >> 5, lane = t & 31;
  const int bs0 = blockIdx.x * SPB;
  const int b = bs0 >> 10;

  for (int r = t; r < 64; r += 128) {
    int sl = r / KN, kk = r % KN;
    int q = bs0 + sl;
    int idx = ball[(size_t)q * KN + kk];
    float nx = newXyz[q * 3 + 0], ny = newXyz[q * 3 + 1], nz = newXyz[q * 3 + 2];
    const float* p = xyz + (size_t)(b * NN + idx) * 3;
    const float* f = feat + (size_t)(b * NN + idx) * 6;
    _Float16* row = bufA + r * 128;
    row[0] = (_Float16)(p[0] - nx);
    row[1] = (_Float16)(p[1] - ny);
    row[2] = (_Float16)(p[2] - nz);
#pragma unroll
    for (int c = 0; c < 6; ++c) row[3 + c] = (_Float16)f[c];
#pragma unroll
    for (int c = 9; c < CP0; ++c) row[c] = (_Float16)0.f;
  }
  __syncthreads();

  mlp_layer<CP0, CO0>(Wall, bias0, bufA, bufB, Wb, t, wave, lane);
  mlp_layer<CP1, CO1>(Wall + WOFF1, bias1, bufB, bufA, Wb, t, wave, lane);
  mlp_layer<CP2, CO2>(Wall + WOFF2, bias2, bufA, bufB, Wb, t, wave, lane);

  for (int o = t; o < SPB * 128; o += 128) {
    int sl = o >> 7, c = o & 127;
    float m = -1e30f;
#pragma unroll 4
    for (int kk = 0; kk < KN; ++kk) {
      float v = (float)bufB[(sl * KN + kk) * 128 + c];
      m = v > m ? v : m;
    }
    int q = bs0 + sl;
    outFeat[(size_t)q * 768 + c0 + c] = m;
    xcur[(size_t)q * 128 + c] = m;
    xf16[(size_t)q * 128 + c] = (_Float16)m;
  }
}

// ---------------------------------------------------------------------------
// Generic f16 GEMM, f32 accumulate: C = act(A(MxK) @ W + bias), W is [N][K].
// block = 128 (4 waves): 64x64 tile. Double-buffered async global->LDS
// pipeline: issue tile i+1, s_wait_asynccnt 4 (tile i complete), compute.
// ---------------------------------------------------------------------------
__global__ void gemm16_kernel(const _Float16* __restrict__ A,
                              const _Float16* __restrict__ WT,
                              const float* __restrict__ bias,
                              float* __restrict__ Cf, _Float16* __restrict__ Ch,
                              int M, int N, int Kd, int relu) {
  __shared__ __align__(16) _Float16 At[2][64 * 32];
  __shared__ __align__(16) _Float16 Wt[2][64 * 32];
  const int t = threadIdx.x, wave = t >> 5, lane = t & 31;
  const int m0 = blockIdx.x * 64;
  const int n0 = blockIdx.y * 64;
  v8f acc[4];
#pragma unroll
  for (int i = 0; i < 4; ++i) acc[i] = vzero8();

  const unsigned long long abase = (unsigned long long)(size_t)A;
  const unsigned long long wbase = (unsigned long long)(size_t)WT;

  auto issue_tile = [&](int ks, int ib) {
#pragma unroll
    for (int u = 0; u < 2; ++u) {
      int i = t + u * 128;                 // 256 16B chunks per 64x32 tile
      int r = i >> 2, sg = (i & 3) * 8;
      async_b128(&At[ib][r * 32 + sg],
                 (unsigned)(((size_t)(m0 + r) * Kd + ks + sg) * 2), abase);
      async_b128(&Wt[ib][r * 32 + sg],
                 (unsigned)(((size_t)(n0 + r) * Kd + ks + sg) * 2), wbase);
    }
  };

  issue_tile(0, 0);
  int ib = 0;
  for (int ks = 0; ks < Kd; ks += 32, ib ^= 1) {
    if (ks + 32 < Kd) {
      issue_tile(ks + 32, ib ^ 1);
      asm volatile("s_wait_asynccnt 0x4" ::: "memory");  // tile ks complete
    } else {
      asm volatile("s_wait_asynccnt 0x0" ::: "memory");
    }
    __syncthreads();
    v16h a = frag_A(At[ib], 32, wave * 16, 0, lane);
#pragma unroll
    for (int nt = 0; nt < 4; ++nt) {
      v16h bf = frag_Bn(Wt[ib], 32, 0, nt * 16, lane);
      acc[nt] = wmma_f16(a, bf, acc[nt]);
    }
    __syncthreads();
  }

  const int rb = ((lane >> 4) << 3);
#pragma unroll
  for (int nt = 0; nt < 4; ++nt) {
    int col = n0 + nt * 16 + (lane & 15);
    float bv = bias ? bias[col] : 0.f;
#pragma unroll
    for (int r8 = 0; r8 < 8; ++r8) {
      int row = m0 + wave * 16 + rb + r8;
      float v = acc[nt][r8] + bv;
      if (relu) v = v > 0.f ? v : 0.f;
      size_t o = (size_t)row * N + col;
      if (Cf) Cf[o] = v;
      if (Ch) Ch[o] = (_Float16)v;
    }
  }
}

// ---------------------------------------------------------------------------
// Fused attention over packed QKV (B,S,384): one block per (b, head, q-tile).
// scores(16x1024) fp32 in LDS -> register softmax -> f16 probs packed in
// place -> att @ vT via WMMA.
// ---------------------------------------------------------------------------
__global__ void attn_kernel(const _Float16* __restrict__ qkv,
                            const _Float16* __restrict__ vT,
                            _Float16* __restrict__ ob) {
  __shared__ float sc[16 * 1024];  // 64 KB
  const int t = threadIdx.x, wave = t >> 5, lane = t & 31;
  const int blk = blockIdx.x;               // B*H*64
  const int qt = blk & 63, h = (blk >> 6) & 3, b = blk >> 8;
  const int q0 = qt * 16;
  const float scale = 0.17677669529663687f; // 1/sqrt(32)

  v16h qf;
  {
    int r = q0 + (lane & 15);
    int kbs = (lane >> 4) << 3;
    const _Float16* p = qkv + ((size_t)(b * SS + r)) * 384 + h * 32 + kbs;
    ld8(qf, 0, p);
    ld8(qf, 8, p + 16);
  }

  for (int jt = wave; jt < 64; jt += 4) {
    v16h kf;
    {
      int n = jt * 16 + (lane & 15);
      int kbs = (lane >> 4) << 3;
      const _Float16* p = qkv + ((size_t)(b * SS + n)) * 384 + 128 + h * 32 + kbs;
      ld8(kf, 0, p);
      ld8(kf, 8, p + 16);
    }
    v8f acc = vzero8();
    acc = wmma_f16(qf, kf, acc);
    const int rb = (lane >> 4) << 3;
#pragma unroll
    for (int r8 = 0; r8 < 8; ++r8)
      sc[(rb + r8) * 1024 + jt * 16 + (lane & 15)] = acc[r8] * scale;
  }
  __syncthreads();

  // Register softmax (32 values/lane), then store f16 probs packed in place.
  _Float16* scH = (_Float16*)sc;  // row r occupies scH[r*2048 .. r*2048+1024)
  for (int r = wave; r < 16; r += 4) {
    float ev[32];
    float mx = -1e30f;
#pragma unroll
    for (int i = 0; i < 32; ++i) {
      ev[i] = sc[r * 1024 + lane + i * 32];
      mx = fmaxf(mx, ev[i]);
    }
#pragma unroll
    for (int m = 16; m; m >>= 1) mx = fmaxf(mx, __shfl_xor(mx, m, 32));
    float sm = 0.f;
#pragma unroll
    for (int i = 0; i < 32; ++i) { ev[i] = __expf(ev[i] - mx); sm += ev[i]; }
#pragma unroll
    for (int m = 16; m; m >>= 1) sm += __shfl_xor(sm, m, 32);
    float inv = 1.f / sm;
#pragma unroll
    for (int i = 0; i < 32; ++i)
      scH[r * 2048 + lane + i * 32] = (_Float16)(ev[i] * inv);
  }
  __syncthreads();

  // O(16x32) = att(16x1024) @ v(1024x32); vT gives contiguous B frags.
  if (wave < 2) {
    const int nt = wave;
    v8f acc = vzero8();
    for (int ks = 0; ks < 1024; ks += 32) {
      v16h a = frag_A(scH, 2048, 0, ks, lane);
      v16h bf;
      {
        int n = nt * 16 + (lane & 15);
        int kbs = ks + ((lane >> 4) << 3);
        const _Float16* p = vT + ((size_t)((b * 4 + h) * 32 + n)) * SS + kbs;
        ld8(bf, 0, p);
        ld8(bf, 8, p + 16);
      }
      acc = wmma_f16(a, bf, acc);
    }
    const int rb = (lane >> 4) << 3;
#pragma unroll
    for (int r8 = 0; r8 < 8; ++r8) {
      int qr = q0 + rb + r8;
      ob[((size_t)(b * SS + qr)) * DMODEL + h * 32 + nt * 16 + (lane & 15)] =
          (_Float16)acc[r8];
    }
  }
}

// ---------------------------------------------------------------------------
// Residual + LayerNorm (d=128): one wave32 per row, 4 elems per lane.
// ---------------------------------------------------------------------------
__global__ void ln_kernel(const float* __restrict__ x, const float* __restrict__ dlt,
                          const float* __restrict__ g, const float* __restrict__ be,
                          float* __restrict__ yo, _Float16* __restrict__ yh) {
  const int row = blockIdx.x * 8 + (threadIdx.x >> 5);
  const int lane = threadIdx.x & 31;
  const float* xr = x + (size_t)row * 128;
  const float* dr = dlt + (size_t)row * 128;
  float v[4];
  float s = 0.f;
#pragma unroll
  for (int i = 0; i < 4; ++i) {
    int c = lane + i * 32;
    v[i] = xr[c] + dr[c];
    s += v[i];
  }
#pragma unroll
  for (int m = 16; m; m >>= 1) s += __shfl_xor(s, m, 32);
  float mean = s * (1.f / 128.f);
  float vs = 0.f;
#pragma unroll
  for (int i = 0; i < 4; ++i) { float d = v[i] - mean; vs += d * d; }
#pragma unroll
  for (int m = 16; m; m >>= 1) vs += __shfl_xor(vs, m, 32);
  float rstd = rsqrtf(vs * (1.f / 128.f) + 1e-5f);
#pragma unroll
  for (int i = 0; i < 4; ++i) {
    int c = lane + i * 32;
    float o = (v[i] - mean) * rstd * g[c] + be[c];
    yo[(size_t)row * 128 + c] = o;
    yh[(size_t)row * 128 + c] = (_Float16)o;
  }
}

__global__ void copy_out_kernel(const float* __restrict__ x, float* __restrict__ outFeat,
                                int c0) {
  int i = blockIdx.x * blockDim.x + threadIdx.x;
  if (i >= BB * SS * 128) return;
  int q = i >> 7, c = i & 127;
  outFeat[(size_t)q * 768 + c0 + c] = x[i];
}

// ---------------------------------------------------------------------------
// Host orchestration.
// ---------------------------------------------------------------------------
extern "C" void kernel_launch(void* const* d_in, const int* in_sizes, int n_in,
                              void* d_out, int out_size, void* d_ws, size_t ws_size,
                              hipStream_t stream) {
  (void)in_sizes; (void)n_in; (void)out_size; (void)ws_size;
  const float* xyz  = (const float*)d_in[0];
  const float* feat = (const float*)d_in[1];
  float* outXyz  = (float*)d_out;
  float* outFeat = outXyz + (size_t)BB * SS * 3;

  char* w = (char*)d_ws;
  size_t off = 0;
  auto take = [&](size_t bytes) -> char* {
    char* p = w + off;
    off += (bytes + 255) & ~(size_t)255;
    return p;
  };
  const size_t M = (size_t)BB * SS;
  int*       ball  = (int*)take(M * 64 * sizeof(int));
  _Float16*  convW = (_Float16*)take(40960 * sizeof(_Float16));
  _Float16*  tw    = (_Float16*)take(196608 * sizeof(_Float16));
  float*     xA    = (float*)take(M * 128 * sizeof(float));
  float*     xB    = (float*)take(M * 128 * sizeof(float));
  _Float16*  xh    = (_Float16*)take(M * 128 * sizeof(_Float16));
  _Float16*  qkvh  = (_Float16*)take(M * 384 * sizeof(_Float16));
  _Float16*  vt    = (_Float16*)take(M * 128 * sizeof(_Float16));
  _Float16*  ah    = (_Float16*)take(M * 128 * sizeof(_Float16));
  float*     obuf  = (float*)take(M * 128 * sizeof(float));
  _Float16*  ffh   = (_Float16*)take(M * 512 * sizeof(_Float16));

  const int OFF_WQ = 0, OFF_WK = 16384, OFF_WV = 32768, OFF_WO = 49152;
  const int OFF_W1 = 65536, OFF_W2 = 131072;

  fps_kernel<<<BB, 256, 0, stream>>>(xyz, outXyz);

  static const float radii[3] = {0.1f, 0.2f, 0.4f};
  static const int   Ksz[3]   = {16, 32, 64};
  static const int   cins[3][3]  = {{9, 64, 64},  {9, 128, 128}, {9, 128, 128}};
  static const int   cpads[3][3] = {{32, 64, 64}, {32, 128, 128}, {32, 128, 128}};
  static const int   couts[3][3] = {{64, 64, 128}, {128, 128, 128}, {128, 128, 128}};

  for (int s = 0; s < 3; ++s) {
    const int base = 2 + 30 * s;
    const int Kn = Ksz[s];

    ball_kernel<<<(int)((M + 255) / 256), 256, 0, stream>>>(
        xyz, outXyz, ball, radii[s] * radii[s], Kn);

    int woff[3]; int acc = 0;
    for (int l = 0; l < 3; ++l) { woff[l] = acc; acc += cpads[s][l] * couts[s][l]; }
    for (int l = 0; l < 3; ++l) {
      int n = cpads[s][l] * couts[s][l];
      conv_w_kernel<<<(n + 255) / 256, 256, 0, stream>>>(
          (const float*)d_in[base + 3 + l], convW + woff[l],
          cins[s][l], cpads[s][l], couts[s][l]);
    }

    if (s == 0) {
      conv_mlp_kernel<16, 4, 32, 64, 64, 64, 64, 128, 2048, 6144>
          <<<(int)(M / 4), 128, 0, stream>>>(
              xyz, feat, outXyz, ball, convW,
              (const float*)d_in[base + 0], (const float*)d_in[base + 1],
              (const float*)d_in[base + 2], outFeat, xA, xh, 0);
    } else if (s == 1) {
      conv_mlp_kernel<32, 2, 32, 128, 128, 128, 128, 128, 4096, 20480>
          <<<(int)(M / 2), 128, 0, stream>>>(
              xyz, feat, outXyz, ball, convW,
              (const float*)d_in[base + 0], (const float*)d_in[base + 1],
              (const float*)d_in[base + 2], outFeat, xA, xh, 256);
    } else {
      conv_mlp_kernel<64, 1, 32, 128, 128, 128, 128, 128, 4096, 20480>
          <<<(int)M, 128, 0, stream>>>(
              xyz, feat, outXyz, ball, convW,
              (const float*)d_in[base + 0], (const float*)d_in[base + 1],
              (const float*)d_in[base + 2], outFeat, xA, xh, 512);
    }

    float* xc = xA; float* xn = xB;
    for (int l = 0; l < 2; ++l) {
      const int lb = base + 6 + 12 * l;
      // {W1,W2,Wk,Wo,Wq,Wv,b1,b2,ln1_b,ln1_g,ln2_b,ln2_g} -> transpose-convert
      cvt_t_kernel<<<64, 256, 0, stream>>>((const float*)d_in[lb + 4], tw + OFF_WQ, 128, 128);
      cvt_t_kernel<<<64, 256, 0, stream>>>((const float*)d_in[lb + 2], tw + OFF_WK, 128, 128);
      cvt_t_kernel<<<64, 256, 0, stream>>>((const float*)d_in[lb + 5], tw + OFF_WV, 128, 128);
      cvt_t_kernel<<<64, 256, 0, stream>>>((const float*)d_in[lb + 3], tw + OFF_WO, 128, 128);
      cvt_t_kernel<<<256, 256, 0, stream>>>((const float*)d_in[lb + 0], tw + OFF_W1, 128, 512);
      cvt_t_kernel<<<256, 256, 0, stream>>>((const float*)d_in[lb + 1], tw + OFF_W2, 512, 128);

      dim3 g384((int)(M / 64), 6), g128((int)(M / 64), 2), g512((int)(M / 64), 8);
      // Fused QKV projection: one pass over xh, N = 384.
      gemm16_kernel<<<g384, 128, 0, stream>>>(xh, tw + OFF_WQ, nullptr, nullptr, qkvh,
                                              (int)M, 384, 128, 0);
      vtrans_kernel<<<(int)((M * 128 + 255) / 256), 256, 0, stream>>>(qkvh, vt);
      attn_kernel<<<BB * 4 * (SS / 16), 128, 0, stream>>>(qkvh, vt, ah);
      gemm16_kernel<<<g128, 128, 0, stream>>>(ah, tw + OFF_WO, nullptr, obuf, nullptr,
                                              (int)M, 128, 128, 0);
      ln_kernel<<<(int)(M / 8), 256, 0, stream>>>(
          xc, obuf, (const float*)d_in[lb + 9], (const float*)d_in[lb + 8], xn, xh);
      { float* t_ = xc; xc = xn; xn = t_; }
      gemm16_kernel<<<g512, 128, 0, stream>>>(xh, tw + OFF_W1,
                                              (const float*)d_in[lb + 6], nullptr, ffh,
                                              (int)M, 512, 128, 1);
      gemm16_kernel<<<g128, 128, 0, stream>>>(ffh, tw + OFF_W2,
                                              (const float*)d_in[lb + 7], obuf, nullptr,
                                              (int)M, 128, 512, 0);
      ln_kernel<<<(int)(M / 8), 256, 0, stream>>>(
          xc, obuf, (const float*)d_in[lb + 11], (const float*)d_in[lb + 10], xn, xh);
      { float* t_ = xc; xc = xn; xn = t_; }
    }
    copy_out_kernel<<<(int)((M * 128 + 255) / 256), 256, 0, stream>>>(
        xc, outFeat, s * 256 + 128);
  }
}
